// PatchEmbedding_90159953478038
// MI455X (gfx1250) — matrix operand order repaired and checked by reading, not compiled, not run on hardware
//
#include <hip/hip_runtime.h>
#include <hip/hip_bf16.h>
#include <stdint.h>

// ---------------------------------------------------------------------------
// PatchEmbedding fused: LN(768) -> [32768x768]x[768x1024] GEMM (bf16 hi/lo
// split, fp32 accumulate via v_wmma_f32_16x16x32_bf16) -> +bias -> LN(1024)
// ---------------------------------------------------------------------------

typedef __attribute__((ext_vector_type(16))) __bf16 v16bf;
typedef __attribute__((ext_vector_type(8)))  float  v8f;

#define PD    768
#define DD    1024
#define KT    24          // PD / 32 k-steps
#define MBLK  64          // rows per workgroup
#define MT    4           // MBLK / 16 m-tiles
#define ROWS  32768       // 32 * 1024
#define LN_EPS 1e-5f

// LDS layout (dynamic shared memory)
#define AFRAG_BYTES (MT * KT * 2048)             // 196608: A fragments hi+lo
#define OFF_PSUM    (AFRAG_BYTES)                // 64 rows x 32 waves partial sums
#define OFF_PSQ     (OFF_PSUM + MBLK * 32 * 4)
#define OFF_MU      (OFF_PSQ  + MBLK * 32 * 4)
#define OFF_RS      (OFF_MU   + MBLK * 4)
#define SMEM_BYTES  (OFF_RS   + MBLK * 4)        // 213504 bytes

union Frag32B { uint4 q[2]; v16bf v; };

// ---------------------------------------------------------------------------
// Prep: swizzle W (fp32, row-major [768][1024]) into fragment-ready bf16
// hi/lo blocks.  Block (kt, nt) = 2KB: hi at lane*32B (16 bf16), lo at
// 1KB + lane*32B.  B-fragment element layout (dense 16-bit B, 32x16):
//   frag lane fl: n = nt*16 + (fl&15), half = fl>>4, element e -> k = half*16+e
// ---------------------------------------------------------------------------
__global__ void __launch_bounds__(256)
wprep_kernel(const float* __restrict__ W, __bf16* __restrict__ ws) {
    const int wid  = (blockIdx.x * blockDim.x + threadIdx.x) >> 5;  // 0..1535
    const int l    = threadIdx.x & 31;
    const int kt   = wid >> 6;          // 0..23
    const int nt   = wid & 63;          // 0..63
    const int n    = nt * 16 + (l & 15);
    const int half = l >> 4;

    const float* wp = W + (size_t)(kt * 32 + half * 16) * DD + n;
    __bf16* hi = ws + (size_t)(kt * 64 + nt) * 1024 + l * 16;   // 1024 bf16/block
    __bf16* lo = hi + 512;

    #pragma unroll
    for (int e = 0; e < 16; ++e) {
        float v  = wp[e * DD];
        __bf16 h = (__bf16)v;                 // hi (RNE)
        hi[e] = h;
        lo[e] = (__bf16)(v - (float)h);       // residual
    }
}

// ---------------------------------------------------------------------------
// Fused main kernel. 1024 threads = 32 waves. Wave w owns N-tiles {2w,2w+1}
// (cols [w*32, w*32+32)) across all 4 M-tiles (64 rows).
// ---------------------------------------------------------------------------
__global__ void __launch_bounds__(1024, 1)
patchembed_fused_kernel(const float* __restrict__ patches,
                        const float* __restrict__ gpre,
                        const __bf16* __restrict__ wfrag,
                        const float* __restrict__ bias,
                        const float* __restrict__ gpost,
                        float* __restrict__ out) {
    extern __shared__ unsigned char smem[];
    __bf16* afrag = (__bf16*)smem;
    float*  pSum  = (float*)(smem + OFF_PSUM);
    float*  pSq   = (float*)(smem + OFF_PSQ);
    float*  rowMu = (float*)(smem + OFF_MU);
    float*  rowRs = (float*)(smem + OFF_RS);

    const int tid  = threadIdx.x;
    const int w    = tid >> 5;           // wave 0..31
    const int l    = tid & 31;           // lane 0..31 (wave32)
    const int row0 = blockIdx.x * MBLK;

    // ---- Stage 1: pre-LayerNorm + bf16 hi/lo split into A-fragment layout --
    // A fragment (this code's contiguous k order, 16-bit A 16x32): for k%32 == l:
//   half = l>>4, e = l&15, fragLane = half*16 + m
    float gp[KT];
    #pragma unroll
    for (int c = 0; c < KT; ++c) gp[c] = gpre[c * 32 + l];

    const int ahalf = l >> 4;
    const int ae    = l & 15;

    for (int rr = 0; rr < 2; ++rr) {
        const int lr = w * 2 + rr;                       // local row 0..63
        const float* xp = patches + (size_t)(row0 + lr) * PD + l;
        float x[KT];
        float s = 0.f, q = 0.f;
        #pragma unroll
        for (int c = 0; c < KT; ++c) {
            x[c] = xp[c * 32];
            s += x[c];
            q += x[c] * x[c];
        }
        #pragma unroll
        for (int o = 16; o > 0; o >>= 1) {
            s += __shfl_xor(s, o, 32);
            q += __shfl_xor(q, o, 32);
        }
        const float mu = s * (1.0f / PD);
        const float rs = rsqrtf(q * (1.0f / PD) - mu * mu + LN_EPS);

        const int mb = lr >> 4, ml = lr & 15;
        const int flane = ahalf * 16 + ml;
        __bf16* hi = afrag + (size_t)(mb * KT) * 1024 + flane * 16 + ae;
        #pragma unroll
        for (int c = 0; c < KT; ++c) {
            float v  = (x[c] - mu) * rs * gp[c];
            __bf16 h = (__bf16)v;
            hi[c * 1024]       = h;                      // hi half of block c
            hi[c * 1024 + 512] = (__bf16)(v - (float)h); // lo half of block c
        }
    }
    __syncthreads();

    // ---- Stage 2: WMMA GEMM, split-bf16 (hh + hl + lh), fp32 accumulate ----
    v8f acc[MT][2];
    #pragma unroll
    for (int mb = 0; mb < MT; ++mb)
        #pragma unroll
        for (int t = 0; t < 2; ++t)
            acc[mb][t] = (v8f){0.f,0.f,0.f,0.f,0.f,0.f,0.f,0.f};

    const int nt0 = w * 2;
    for (int kt = 0; kt < KT; ++kt) {
        v16bf ah[MT], al[MT];
        #pragma unroll
        for (int mb = 0; mb < MT; ++mb) {
            const uint4* ap = (const uint4*)(afrag + (size_t)(mb * KT + kt) * 1024 + l * 16);
            Frag32B fh, fl;
            fh.q[0] = ap[0];  fh.q[1] = ap[1];     // hi: 32B per lane
            fl.q[0] = ap[64]; fl.q[1] = ap[65];    // lo: +1024B
            ah[mb] = fh.v;  al[mb] = fl.v;
        }
        #pragma unroll
        for (int t = 0; t < 2; ++t) {
            const uint4* bp = (const uint4*)(wfrag + (size_t)(kt * 64 + nt0 + t) * 1024 + l * 16);
            Frag32B bh, bl;
            bh.q[0] = bp[0];  bh.q[1] = bp[1];
            bl.q[0] = bp[64]; bl.q[1] = bp[65];
            #pragma unroll
            for (int mb = 0; mb < MT; ++mb) {
                acc[mb][t] = __builtin_amdgcn_wmma_f32_16x16x32_bf16(
                    false, ah[mb], false, bh.v, (short)0, acc[mb][t], false, false);
                acc[mb][t] = __builtin_amdgcn_wmma_f32_16x16x32_bf16(
                    false, ah[mb], false, bl.v, (short)0, acc[mb][t], false, false);
                acc[mb][t] = __builtin_amdgcn_wmma_f32_16x16x32_bf16(
                    false, al[mb], false, bh.v, (short)0, acc[mb][t], false, false);
            }
        }
    }

    // ---- Stage 3: bias + post-LayerNorm from registers ---------------------
    // C/D layout: lane l holds col n = (l&15); VGPR r holds row r + 8*(l>>4)
    const int c0 = w * 32 + (l & 15);
    const int hC = l >> 4;
    const float b0 = bias[c0],  b1 = bias[c0 + 16];
    const float g0 = gpost[c0], g1 = gpost[c0 + 16];

    #pragma unroll
    for (int mb = 0; mb < MT; ++mb)
        #pragma unroll
        for (int r = 0; r < 8; ++r) {
            acc[mb][0][r] += b0;
            acc[mb][1][r] += b1;
        }

    // per-wave partial sums over its 32 columns, for every one of the 64 rows
    #pragma unroll
    for (int mb = 0; mb < MT; ++mb) {
        #pragma unroll
        for (int r = 0; r < 8; ++r) {
            float v0 = acc[mb][0][r], v1 = acc[mb][1][r];
            float s = v0 + v1, q = v0 * v0 + v1 * v1;
            #pragma unroll
            for (int o = 8; o > 0; o >>= 1) {   // reduce 16 lanes within half
                s += __shfl_xor(s, o, 32);
                q += __shfl_xor(q, o, 32);
            }
            if ((l & 15) == 0) {
                const int row = mb * 16 + r + 8 * hC;
                pSum[row * 32 + w] = s;
                pSq [row * 32 + w] = q;
            }
        }
    }
    __syncthreads();

    // combine 32 wave-partials per row -> mean, rstd
    {
        const int row = tid >> 4;   // 0..63, 16 threads per row
        const int j   = tid & 15;
        float s = pSum[row * 32 + j] + pSum[row * 32 + j + 16];
        float q = pSq [row * 32 + j] + pSq [row * 32 + j + 16];
        #pragma unroll
        for (int o = 8; o > 0; o >>= 1) {
            s += __shfl_xor(s, o, 32);
            q += __shfl_xor(q, o, 32);
        }
        if (j == 0) {
            const float mu = s * (1.0f / DD);
            rowMu[row] = mu;
            rowRs[row] = rsqrtf(q * (1.0f / DD) - mu * mu + LN_EPS);
        }
    }
    __syncthreads();

    // normalize + store
    #pragma unroll
    for (int mb = 0; mb < MT; ++mb) {
        #pragma unroll
        for (int r = 0; r < 8; ++r) {
            const int row = mb * 16 + r + 8 * hC;
            const float mu = rowMu[row], rs = rowRs[row];
            const size_t base = (size_t)(row0 + row) * DD;
            out[base + c0]      = (acc[mb][0][r] - mu) * rs * g0;
            out[base + c0 + 16] = (acc[mb][1][r] - mu) * rs * g1;
        }
    }
}

// ---------------------------------------------------------------------------
extern "C" void kernel_launch(void* const* d_in, const int* in_sizes, int n_in,
                              void* d_out, int out_size, void* d_ws, size_t ws_size,
                              hipStream_t stream) {
    const float* patches = (const float*)d_in[0];   // [32,1024,768]
    const float* gpre    = (const float*)d_in[1];   // [768]
    const float* W       = (const float*)d_in[2];   // [768,1024]
    const float* bias    = (const float*)d_in[3];   // [1024]
    const float* gpost   = (const float*)d_in[4];   // [1024]
    float* out           = (float*)d_out;           // [32,1024,1024]
    __bf16* wws          = (__bf16*)d_ws;           // needs 3 MB (hi/lo W frags)

    // allow large dynamic LDS (320KB/WG is architecturally legal on CDNA5)
    (void)hipFuncSetAttribute((const void*)patchembed_fused_kernel,
                              hipFuncAttributeMaxDynamicSharedMemorySize,
                              SMEM_BYTES);

    // 1) swizzle W into WMMA B-fragment hi/lo layout (24*64 blocks, 1 wave each)
    wprep_kernel<<<192, 256, 0, stream>>>(W, wws);

    // 2) fused LN -> GEMM -> LN: 512 workgroups of 64 rows
    patchembed_fused_kernel<<<ROWS / MBLK, 1024, SMEM_BYTES, stream>>>(
        patches, gpre, wws, bias, gpost, out);
}
